// ThreeConvGlobal_57157424775215
// MI455X (gfx1250) — compile-verified, matrix operand order checked by reading, not run on hardware
//
#include <hip/hip_runtime.h>
#include <hip/hip_bf16.h>

typedef __attribute__((ext_vector_type(16))) __bf16 v16bf;
typedef __attribute__((ext_vector_type(8)))  __bf16 v8bf;
typedef __attribute__((ext_vector_type(8)))  float  v8f;

#define KNN_SLICES 8

// ---------------------------------------------------------------------------
// top-6 sorted insert (ascending; bd[5] is current worst)
// ---------------------------------------------------------------------------
__device__ __forceinline__ void topk_insert(float (&bd)[6], int (&bi)[6], float d, int id) {
    if (d < bd[5]) {
        bd[5] = d; bi[5] = id;
#pragma unroll
        for (int t = 5; t > 0; --t) {
            if (bd[t] < bd[t - 1]) {
                float tf = bd[t]; bd[t] = bd[t - 1]; bd[t - 1] = tf;
                int   ti = bi[t]; bi[t] = bi[t - 1]; bi[t - 1] = ti;
            }
        }
    }
}

// ---------------------------------------------------------------------------
// Scatter-add of x (16ch) over edges + self loops; counts too.
// ---------------------------------------------------------------------------
__global__ void scatter_x_kernel(const float* __restrict__ X, const int* __restrict__ ei,
                                 int E, int n, float* __restrict__ acc, float* __restrict__ cnt) {
    int e = blockIdx.x * blockDim.x + threadIdx.x;
    if (e >= E + n) return;
    int s, d;
    if (e < E) { s = ei[e]; d = ei[E + e]; } else { s = e - E; d = s; }
    const float4* xp = (const float4*)(X + (size_t)s * 16);
    float4 a = xp[0], b = xp[1], c = xp[2], g = xp[3];
    float vals[16] = {a.x,a.y,a.z,a.w,b.x,b.y,b.z,b.w,c.x,c.y,c.z,c.w,g.x,g.y,g.z,g.w};
#pragma unroll
    for (int k = 0; k < 16; ++k) atomicAdd(&acc[(size_t)d * 16 + k], vals[k]);
    atomicAdd(&cnt[d], 1.0f);
}

// Scatter-add of cat0 = [x1(16) | y1(32)] over the same edge list.
__global__ void scatter_cat_kernel(const float* __restrict__ X1, const float* __restrict__ Y1,
                                   const int* __restrict__ ei, int E, int n,
                                   float* __restrict__ acc) {
    int e = blockIdx.x * blockDim.x + threadIdx.x;
    if (e >= E + n) return;
    int s, d;
    if (e < E) { s = ei[e]; d = ei[E + e]; } else { s = e - E; d = s; }
#pragma unroll
    for (int k = 0; k < 16; ++k) atomicAdd(&acc[(size_t)d * 48 + k], X1[(size_t)s * 16 + k]);
#pragma unroll
    for (int k = 0; k < 32; ++k) atomicAdd(&acc[(size_t)d * 48 + 16 + k], Y1[(size_t)s * 32 + k]);
}

// ---------------------------------------------------------------------------
// FeaSt finalize: relu(mean @ W + b).  (heads==1 => softmax == 1, linearity)
// ---------------------------------------------------------------------------
template <int IN, int OUT>
__global__ void feast_finalize_kernel(const float* __restrict__ acc, const float* __restrict__ cnt,
                                      const float* __restrict__ W, const float* __restrict__ b,
                                      float* __restrict__ out, int n) {
    int i = blockIdx.x * blockDim.x + threadIdx.x;
    if (i >= n) return;
    float inv = 1.0f / fmaxf(cnt[i], 1.0f);
    float m[IN];
#pragma unroll
    for (int k = 0; k < IN; ++k) m[k] = acc[(size_t)i * IN + k] * inv;
    for (int o = 0; o < OUT; ++o) {
        float h = b[o];
#pragma unroll
        for (int k = 0; k < IN; ++k) h += m[k] * W[k * OUT + o];
        out[(size_t)i * OUT + o] = fmaxf(h, 0.0f);
    }
}

// ---------------------------------------------------------------------------
// bf16-pad features to K=32 + squared norms (norms from the *rounded* values
// so the WMMA dot and the norm terms are consistent -> diagonal ~ 0).
// ---------------------------------------------------------------------------
__global__ void prep_bf16_sq_kernel(const float* __restrict__ X, int C,
                                    __bf16* __restrict__ Xb, float* __restrict__ sq, int n) {
    int i = blockIdx.x * blockDim.x + threadIdx.x;
    if (i >= n) return;
    float s = 0.0f;
    for (int c = 0; c < 32; ++c) {
        float v = (c < C) ? X[(size_t)i * C + c] : 0.0f;
        __bf16 vb = (__bf16)v;
        float vr = (float)vb;
        s += vr * vr;
        Xb[(size_t)i * 32 + c] = vb;
    }
    sq[i] = s;
}

// ---------------------------------------------------------------------------
// Fused WMMA kNN (top-6 smallest squared distances), column-sliced.
// One wave handles a 16-query-row tile against one column slice.
// D = A(16x32, candidate rows) x B(32x16, query cols); per ISA layout, lane l
// of the f32 D tile holds N = l&15 (query) and M = v + 8*(l>>4) (candidate),
// so the lane pair (l, l+16) owns one query row. Each lane keeps a private
// sorted top-6; the pair is merged with __shfl_xor at the end.
// ---------------------------------------------------------------------------
__global__ __launch_bounds__(32)
void knn_partial_kernel(const __bf16* __restrict__ Xb, const float* __restrict__ sq,
                        float* __restrict__ part_d, int* __restrict__ part_i, int n) {
    const int lane = threadIdx.x;
    const int half = lane >> 4;
    const int r    = lane & 15;
    const int q    = blockIdx.x * 16 + r;
    const int ntiles = n >> 4;
    const int slice  = blockIdx.y;
    const int t0 = (slice * ntiles) / KNN_SLICES;
    const int t1 = ((slice + 1) * ntiles) / KNN_SLICES;

    // Query fragment (B operand, 32x16 bf16): lane holds query column q.
    // 16-bit operand layout: lanes<16 carry K={0..7,16..23}, lanes>=16 K={8..15,24..31}.
    v16bf Bfrag;
    {
        const v8bf lo = *(const v8bf*)(Xb + (size_t)q * 32 + half * 8);
        const v8bf hi = *(const v8bf*)(Xb + (size_t)q * 32 + 16 + half * 8);
#pragma unroll
        for (int i = 0; i < 8; ++i) { Bfrag[i] = lo[i]; Bfrag[i + 8] = hi[i]; }
    }
    const float sqQ = sq[q];

    float bd[6]; int bi[6];
#pragma unroll
    for (int t = 0; t < 6; ++t) { bd[t] = 3.0e38f; bi[t] = 0; }

    for (int ct = t0; ct < t1; ++ct) {
        const int cbase = ct * 16;
        if (ct + 1 < t1)
            __builtin_prefetch(Xb + (size_t)(cbase + 16 + r) * 32, 0, 0);

        // Candidate fragment (A operand, 16x32 bf16): lane holds row cbase+r.
        v16bf Afrag;
        {
            const v8bf lo = *(const v8bf*)(Xb + (size_t)(cbase + r) * 32 + half * 8);
            const v8bf hi = *(const v8bf*)(Xb + (size_t)(cbase + r) * 32 + 16 + half * 8);
#pragma unroll
            for (int i = 0; i < 8; ++i) { Afrag[i] = lo[i]; Afrag[i + 8] = hi[i]; }
        }

        v8f acc = {};
        acc = __builtin_amdgcn_wmma_f32_16x16x32_bf16(false, Afrag, false, Bfrag,
                                                      (short)0, acc, false, false);

        // candidate squared norms for this lane's 8 rows (contiguous)
        const float4 s0 = *(const float4*)(sq + cbase + half * 8);
        const float4 s1 = *(const float4*)(sq + cbase + half * 8 + 4);
        const float scand[8] = {s0.x, s0.y, s0.z, s0.w, s1.x, s1.y, s1.z, s1.w};

#pragma unroll
        for (int v = 0; v < 8; ++v) {
            const int cand = cbase + half * 8 + v;
            float d = sqQ + scand[v] - 2.0f * acc[v];
            d = (cand == q) ? 3.0e38f : d;   // exclude self (diag = inf)
            topk_insert(bd, bi, d, cand);
        }
    }

    // Merge the two half-lane lists for this query row.
#pragma unroll
    for (int t = 0; t < 6; ++t) {
        float pd = __shfl_xor(bd[t], 16, 32);
        int   pi = __shfl_xor(bi[t], 16, 32);
        topk_insert(bd, bi, pd, pi);
    }
    if (lane < 16) {
        const size_t base = ((size_t)q * KNN_SLICES + slice) * 6;
#pragma unroll
        for (int t = 0; t < 6; ++t) { part_d[base + t] = bd[t]; part_i[base + t] = bi[t]; }
    }
}

__global__ void knn_merge_kernel(const float* __restrict__ part_d, const int* __restrict__ part_i,
                                 int* __restrict__ knn, int n) {
    int i = blockIdx.x * blockDim.x + threadIdx.x;
    if (i >= n) return;
    float bd[6]; int bi[6];
#pragma unroll
    for (int t = 0; t < 6; ++t) { bd[t] = 3.0e38f; bi[t] = 0; }
    const size_t base = (size_t)i * KNN_SLICES * 6;
    for (int s = 0; s < KNN_SLICES * 6; ++s)
        topk_insert(bd, bi, part_d[base + s], part_i[base + s]);
#pragma unroll
    for (int t = 0; t < 6; ++t) knn[(size_t)i * 6 + t] = bi[t];
}

// ---------------------------------------------------------------------------
// EdgeConv MLP: h = relu([xi, xj-xi] @ W1 + b1) @ W2 + b2; max over 6; relu.
// One wave per node (lane = channel), 8 nodes / 256-thread block, weights in LDS.
// ---------------------------------------------------------------------------
template <int IN>
__global__ __launch_bounds__(256)
void edge_mlp_kernel(const float* __restrict__ X, const int* __restrict__ knn,
                     const float* __restrict__ W1, const float* __restrict__ b1,
                     const float* __restrict__ W2, const float* __restrict__ b2,
                     float* __restrict__ Y, int n) {
    __shared__ float sW1[2 * IN * 32];
    __shared__ float sW2[32 * 32];
    __shared__ float sb1[32], sb2[32];
    __shared__ float sxi[8][IN];
    __shared__ float sxj[8][IN];
    __shared__ float shid[8][32];

    for (int i = threadIdx.x; i < 2 * IN * 32; i += 256) sW1[i] = W1[i];
    for (int i = threadIdx.x; i < 32 * 32; i += 256)     sW2[i] = W2[i];
    if (threadIdx.x < 32) { sb1[threadIdx.x] = b1[threadIdx.x]; sb2[threadIdx.x] = b2[threadIdx.x]; }

    const int wave = threadIdx.x >> 5;
    const int lane = threadIdx.x & 31;
    const int node = blockIdx.x * 8 + wave;
    const bool active = node < n;

    if (active && lane < IN) sxi[wave][lane] = X[(size_t)node * IN + lane];

    float omax = -3.0e38f;
    for (int e = 0; e < 6; ++e) {
        __syncthreads();
        if (active && lane < IN) {
            int j = knn[(size_t)node * 6 + e];
            sxj[wave][lane] = X[(size_t)j * IN + lane];
        }
        __syncthreads();
        float h = sb1[lane];
        if (active) {
#pragma unroll
            for (int k = 0; k < IN; ++k) {
                float xi = sxi[wave][k];
                float dj = sxj[wave][k] - xi;
                h += xi * sW1[k * 32 + lane] + dj * sW1[(IN + k) * 32 + lane];
            }
        }
        shid[wave][lane] = fmaxf(h, 0.0f);
        __syncthreads();
        float o = sb2[lane];
        if (active) {
#pragma unroll
            for (int k = 0; k < 32; ++k) o += shid[wave][k] * sW2[k * 32 + lane];
        }
        omax = fmaxf(omax, o);
    }
    if (active) Y[(size_t)node * 32 + lane] = fmaxf(omax, 0.0f);
}

// ---------------------------------------------------------------------------
// Fused MLP head: cat[x3(64), y2(32)] -> 96 -> 32 -> 8 -> 1 (sigmoid).
// One wave per node, all weights (~50 KB) staged in LDS (320 KB/WGP).
// ---------------------------------------------------------------------------
__global__ __launch_bounds__(256)
void mlp_head_kernel(const float* __restrict__ X3, const float* __restrict__ Y2,
                     const float* __restrict__ W1, const float* __restrict__ B1,
                     const float* __restrict__ W2, const float* __restrict__ B2,
                     const float* __restrict__ W3, const float* __restrict__ B3,
                     const float* __restrict__ WO, const float* __restrict__ BO,
                     float* __restrict__ out, int n) {
    __shared__ float sW1[96 * 96];
    __shared__ float sW2[96 * 32];
    __shared__ float sW3[32 * 8];
    __shared__ float sWO[8];
    __shared__ float sB1[96], sB2[32], sB3[8];
    __shared__ float sin_[8][96];
    __shared__ float sh1[8][96];
    __shared__ float sh2[8][32];
    __shared__ float sh3[8][8];

    for (int i = threadIdx.x; i < 96 * 96; i += 256) sW1[i] = W1[i];
    for (int i = threadIdx.x; i < 96 * 32; i += 256) sW2[i] = W2[i];
    for (int i = threadIdx.x; i < 32 * 8; i += 256)  sW3[i] = W3[i];
    for (int i = threadIdx.x; i < 96; i += 256)      sB1[i] = B1[i];
    if (threadIdx.x < 32) sB2[threadIdx.x] = B2[threadIdx.x];
    if (threadIdx.x < 8)  { sB3[threadIdx.x] = B3[threadIdx.x]; sWO[threadIdx.x] = WO[threadIdx.x]; }

    const int wave = threadIdx.x >> 5;
    const int lane = threadIdx.x & 31;
    const int node = blockIdx.x * 8 + wave;
    const bool active = node < n;

    if (active) {
        for (int t = lane; t < 96; t += 32)
            sin_[wave][t] = (t < 64) ? X3[(size_t)node * 64 + t] : Y2[(size_t)node * 32 + (t - 64)];
    }
    __syncthreads();

    if (active) {
#pragma unroll
        for (int t = 0; t < 3; ++t) {
            int o = lane + 32 * t;
            float h = sB1[o];
            for (int k = 0; k < 96; ++k) h += sin_[wave][k] * sW1[k * 96 + o];
            sh1[wave][o] = fmaxf(h, 0.0f);
        }
    }
    __syncthreads();

    if (active) {
        float h = sB2[lane];
        for (int k = 0; k < 96; ++k) h += sh1[wave][k] * sW2[k * 32 + lane];
        sh2[wave][lane] = fmaxf(h, 0.0f);
    }
    __syncthreads();

    if (active && lane < 8) {
        float h = sB3[lane];
#pragma unroll
        for (int k = 0; k < 32; ++k) h += sh2[wave][k] * sW3[k * 8 + lane];
        sh3[wave][lane] = fmaxf(h, 0.0f);
    }
    __syncthreads();

    if (active && lane == 0) {
        float h = BO[0];
#pragma unroll
        for (int k = 0; k < 8; ++k) h += sh3[wave][k] * sWO[k];
        out[node] = 1.0f / (1.0f + __expf(-h));
    }
}

// ---------------------------------------------------------------------------
// Host orchestration
// ---------------------------------------------------------------------------
extern "C" void kernel_launch(void* const* d_in, const int* in_sizes, int n_in,
                              void* d_out, int out_size, void* d_ws, size_t ws_size,
                              hipStream_t stream) {
    (void)n_in; (void)out_size; (void)ws_size;
    const int n = in_sizes[0] / 16;   // x: (N,16)
    const int E = in_sizes[1] / 2;    // edge_index: (2,E)

    const float* x    = (const float*)d_in[0];
    const int*   ei   = (const int*)  d_in[1];
    const float* cW1  = (const float*)d_in[2];
    const float* cb1  = (const float*)d_in[5];
    const float* cW3  = (const float*)d_in[6];
    const float* cb3  = (const float*)d_in[9];
    const float* e1W1 = (const float*)d_in[10];
    const float* e1b1 = (const float*)d_in[11];
    const float* e1W2 = (const float*)d_in[12];
    const float* e1b2 = (const float*)d_in[13];
    const float* e2W1 = (const float*)d_in[14];
    const float* e2b1 = (const float*)d_in[15];
    const float* e2W2 = (const float*)d_in[16];
    const float* e2b2 = (const float*)d_in[17];
    const float* l1W  = (const float*)d_in[18];
    const float* l1b  = (const float*)d_in[19];
    const float* l2W  = (const float*)d_in[20];
    const float* l2b  = (const float*)d_in[21];
    const float* l3W  = (const float*)d_in[22];
    const float* l3b  = (const float*)d_in[23];
    const float* oW   = (const float*)d_in[24];
    const float* ob   = (const float*)d_in[25];

    char* ws = (char*)d_ws;
    auto alloc = [&](size_t bytes) -> char* {
        char* p = ws;
        ws += (bytes + 255) & ~(size_t)255;
        return p;
    };
    float*  acc1   = (float*) alloc((size_t)n * 16 * 4);
    float*  cnt    = (float*) alloc((size_t)n * 4);
    float*  x1     = (float*) alloc((size_t)n * 16 * 4);
    __bf16* Xb1    = (__bf16*)alloc((size_t)n * 32 * 2);
    float*  sq1    = (float*) alloc((size_t)n * 4);
    int*    idx1   = (int*)   alloc((size_t)n * 6 * 4);
    float*  y1     = (float*) alloc((size_t)n * 32 * 4);
    __bf16* Xb2    = (__bf16*)alloc((size_t)n * 32 * 2);
    float*  sq2    = (float*) alloc((size_t)n * 4);
    int*    idx2   = (int*)   alloc((size_t)n * 6 * 4);
    float*  y2     = (float*) alloc((size_t)n * 32 * 4);
    float*  acc3   = (float*) alloc((size_t)n * 48 * 4);
    float*  x3     = (float*) alloc((size_t)n * 64 * 4);
    float*  part_d = (float*) alloc((size_t)n * KNN_SLICES * 6 * 4);
    int*    part_i = (int*)   alloc((size_t)n * KNN_SLICES * 6 * 4);

    hipMemsetAsync(acc1, 0, (size_t)n * 16 * 4, stream);
    hipMemsetAsync(cnt,  0, (size_t)n * 4, stream);
    hipMemsetAsync(acc3, 0, (size_t)n * 48 * 4, stream);

    const int tot = E + n;
    const int TB = 256;

    // FeaSt 1 (heads=1 -> mean + linear)
    scatter_x_kernel<<<(tot + TB - 1) / TB, TB, 0, stream>>>(x, ei, E, n, acc1, cnt);
    feast_finalize_kernel<16, 16><<<(n + TB - 1) / TB, TB, 0, stream>>>(acc1, cnt, cW1, cb1, x1, n);

    // DynamicEdgeConv 1
    prep_bf16_sq_kernel<<<(n + TB - 1) / TB, TB, 0, stream>>>(x1, 16, Xb1, sq1, n);
    knn_partial_kernel<<<dim3(n / 16, KNN_SLICES), 32, 0, stream>>>(Xb1, sq1, part_d, part_i, n);
    knn_merge_kernel<<<(n + TB - 1) / TB, TB, 0, stream>>>(part_d, part_i, idx1, n);
    edge_mlp_kernel<16><<<(n + 7) / 8, TB, 0, stream>>>(x1, idx1, e1W1, e1b1, e1W2, e1b2, y1, n);

    // DynamicEdgeConv 2
    prep_bf16_sq_kernel<<<(n + TB - 1) / TB, TB, 0, stream>>>(y1, 32, Xb2, sq2, n);
    knn_partial_kernel<<<dim3(n / 16, KNN_SLICES), 32, 0, stream>>>(Xb2, sq2, part_d, part_i, n);
    knn_merge_kernel<<<(n + TB - 1) / TB, TB, 0, stream>>>(part_d, part_i, idx2, n);
    edge_mlp_kernel<32><<<(n + 7) / 8, TB, 0, stream>>>(y1, idx2, e2W1, e2b1, e2W2, e2b2, y2, n);

    // FeaSt 3 on cat0 = [x1 | y1]
    scatter_cat_kernel<<<(tot + TB - 1) / TB, TB, 0, stream>>>(x1, y1, ei, E, n, acc3);
    feast_finalize_kernel<48, 64><<<(n + TB - 1) / TB, TB, 0, stream>>>(acc3, cnt, cW3, cb3, x3, n);

    // Head MLP on cat1 = [x3 | y2]
    mlp_head_kernel<<<(n + 7) / 8, TB, 0, stream>>>(x3, y2, l1W, l1b, l2W, l2b, l3W, l3b,
                                                    oW, ob, (float*)d_out, n);
}
